// GATModel_72267119722747
// MI455X (gfx1250) — compile-verified, matrix operand order checked by reading, not run on hardware
//
#include <hip/hip_runtime.h>
#include <hip/hip_bf16.h>

typedef __attribute__((ext_vector_type(16))) _Float16 v16h;
typedef __attribute__((ext_vector_type(8)))  _Float16 v8h;
typedef __attribute__((ext_vector_type(8)))  float    v8f;
typedef __attribute__((ext_vector_type(4)))  unsigned int v4u;
typedef __attribute__((ext_vector_type(4)))  int      v4i;
typedef __attribute__((ext_vector_type(8)))  int      v8i;

#define GAT_ALPHA 0.2f

#if defined(__AMDGCN__) && __has_builtin(__builtin_amdgcn_tensor_load_to_lds)
#define GAT_HAVE_TDM 1
#else
#define GAT_HAVE_TDM 0
#endif

#if GAT_HAVE_TDM
// ---------------------------------------------------------------------------
// Tensor Data Mover: 1D tile copy global->LDS via D# descriptor (ISA ch.8).
// group0: [1:0]=count=1, [63:32]=lds_addr, [120:64]=global_addr, [127:126]=2.
// group1: [17:16]=data_size(2 -> 4B), tensor_dim0=N, tensor_dim1=1,
//         tile_dim0=N (tile_dim1/2 = 0 -> 1D), tensor_dim0_stride=N.
// groups 2/3: zero (<=2D tensor).
// ---------------------------------------------------------------------------
static __device__ __forceinline__ void tdm_load_1d_f32(const float* gsrc,
                                                       float* lds_dst,
                                                       unsigned nelem) {
  const unsigned long long ga = (unsigned long long)gsrc;
  const unsigned lds = (unsigned)(unsigned long long)lds_dst;  // low 32 = LDS
  v4u g0;
  g0[0] = 1u;                                  // count=1 (valid user D#)
  g0[1] = lds;                                 // lds_addr (bytes)
  g0[2] = (unsigned)ga;                        // global_addr[31:0]
  g0[3] = (unsigned)(ga >> 32) | (2u << 30);   // global_addr[56:32] | type=2
  v8i g1;
  g1[0] = (int)(2u << 16);                          // data_size=2 (4 bytes)
  g1[1] = (int)((nelem & 0xFFFFu) << 16);           // tensor_dim0[15:0]
  g1[2] = (int)((nelem >> 16) | (1u << 16));        // dim0[31:16] | dim1=1
  g1[3] = (int)((nelem & 0xFFFFu) << 16);           // tile_dim0 = nelem
  g1[4] = 0;                                        // tile_dim1/2 = 0 (1D)
  g1[5] = (int)nelem;                               // dim0_stride[31:0]
  g1[6] = 0;
  g1[7] = 0;
  const v4i z4 = {0, 0, 0, 0};
#if __clang_major__ >= 23
  const v8i z8 = {0, 0, 0, 0, 0, 0, 0, 0};
  __builtin_amdgcn_tensor_load_to_lds(g0, g1, z4, z4, z8, 0);
#else
  __builtin_amdgcn_tensor_load_to_lds(g0, g1, z4, z4, 0);
#endif
}
#endif  // GAT_HAVE_TDM

// ---------------------------------------------------------------------------
// CDNA5 WMMA fragment I/O (wave32), ISA 7.12.2.
// A 16x32 f16 (MxK): lane L holds M=L%16; hi=L/16 adds +8 to K.
//   f[0..7]  <- K = k0 + 8*hi + (0..7)
//   f[8..15] <- K = k0 + 8*hi + 16 + (0..7)
// Row-major f16 storage -> two aligned 16-byte LDS loads per fragment.
// B 32x16 f16 (KxN) uses the identical mapping with N=L%16 ("column-major"),
// so the same loader on a TRANSPOSED array yields B fragments.
// C/D 16x16 f32: VGPR r holds M = r + 8*(L/16), N = L%16.
// ---------------------------------------------------------------------------
static __device__ __forceinline__ v16h frag_row_f16(const _Float16* src, int ld,
                                                    int row0, int k0, int lane) {
  const _Float16* p = src + (row0 + (lane & 15)) * ld + k0 + ((lane >> 4) << 3);
  v8h lo = *(const v8h*)p;            // ds_load_b128
  v8h hi = *(const v8h*)(p + 16);     // ds_load_b128
  return __builtin_shufflevector(lo, hi, 0, 1, 2, 3, 4, 5, 6, 7, 8, 9, 10, 11,
                                 12, 13, 14, 15);
}

static __device__ __forceinline__ void store_c_row_f16(_Float16* dst, int ld,
                                                       int row0, int col0,
                                                       int lane, v8f c,
                                                       bool do_elu) {
  const int n  = col0 + (lane & 15);
  const int m0 = row0 + ((lane >> 4) << 3);
#pragma unroll
  for (int r = 0; r < 8; ++r) {
    float x = c[r];
    if (do_elu) x = x > 0.f ? x : (__expf(x) - 1.f);
    dst[(m0 + r) * ld + n] = (_Float16)x;
  }
}

// Transposed f16 store of a C tile: the 8 M-values per lane are contiguous in
// the transposed array -> one aligned 16-byte ds_store_b128.
static __device__ __forceinline__ void store_c_tr_f16(_Float16* dstT, int ldT,
                                                      int row0, int col0,
                                                      int lane, v8f c) {
  const int n  = col0 + (lane & 15);
  const int m0 = row0 + ((lane >> 4) << 3);
  v8h h;
#pragma unroll
  for (int r = 0; r < 8; ++r) h[r] = (_Float16)c[r];
  *(v8h*)(dstT + n * ldT + m0) = h;
}

// One GAT attention stage on LDS buffers: scores from avec (global, 256 f32),
// row softmax, Ho = elu(attn @ Wh). Wh comes TRANSPOSED -> direct B fragments.
static __device__ __forceinline__ void gat_attention(
    const float* __restrict__ avec, const _Float16* sWhT, _Float16* sHo,
    _Float16* sAt, float* sSi, float* sSj, int tid, int lane, int wave) {
  if (tid < 64) {
    float si = 0.f, sj = 0.f;
#pragma unroll 4
    for (int d = 0; d < 128; ++d) {
      const float w = (float)sWhT[d * 64 + tid];
      si = fmaf(w, avec[d], si);
      sj = fmaf(w, avec[128 + d], sj);
    }
    sSi[tid] = si;
    sSj[tid] = sj;
  }
  __syncthreads();
  if (tid < 64) {
    const float si = sSi[tid];
    float mx = -1e30f;
    for (int j = 0; j < 64; ++j) {
      float e = si + sSj[j];
      e = e > 0.f ? e : GAT_ALPHA * e;
      mx = fmaxf(mx, e);
    }
    float sum = 0.f;
    float ex[64];
#pragma unroll 8
    for (int j = 0; j < 64; ++j) {
      float e = si + sSj[j];
      e = e > 0.f ? e : GAT_ALPHA * e;
      ex[j] = __expf(e - mx);
      sum += ex[j];
    }
    const float inv = 1.f / sum;
#pragma unroll 8
    for (int j = 0; j < 64; ++j) sAt[tid * 64 + j] = (_Float16)(ex[j] * inv);
  }
  __syncthreads();
  // Ho = elu(attn(64x64) @ Wh(64x128)); wave owns N-strip [16w,16w+16).
  const int n0 = wave * 16;
  const v16h b0 = frag_row_f16(sWhT, 64, n0, 0, lane);
  const v16h b1 = frag_row_f16(sWhT, 64, n0, 32, lane);
#pragma unroll
  for (int mt = 0; mt < 4; ++mt) {
    v8f c = {};
    v16h a0 = frag_row_f16(sAt, 64, mt * 16, 0, lane);
    v16h a1 = frag_row_f16(sAt, 64, mt * 16, 32, lane);
    c = __builtin_amdgcn_wmma_f32_16x16x32_f16(false, a0, false, b0, (short)0,
                                               c, false, false);
    c = __builtin_amdgcn_wmma_f32_16x16x32_f16(false, a1, false, b1, (short)0,
                                               c, false, false);
    store_c_row_f16(sHo, 128, mt * 16, n0, lane, c, /*elu=*/true);
  }
  __syncthreads();
}

__global__ __launch_bounds__(256) void gat_fused_kernel(
    const float* __restrict__ users, const float* __restrict__ W1,
    const float* __restrict__ a1, const float* __restrict__ W2,
    const float* __restrict__ a2, const float* __restrict__ mw1,
    const float* __restrict__ mb1, const float* __restrict__ mw2,
    const float* __restrict__ mb2, float* __restrict__ out_power,
    float* __restrict__ out_delta) {
  __shared__ __align__(16) float    sW2stage[128 * 128];  // 64 KB TDM dest
  __shared__ __align__(16) float    sUsers[64 * 3];       // 768 B TDM dest
  __shared__ __align__(16) _Float16 sWhT[128 * 64];       // 16 KB Wh^T (B op)
  __shared__ __align__(16) _Float16 sHo[64 * 128];        // 16 KB h (A op)
  __shared__ __align__(16) _Float16 sAt[64 * 64];         //  8 KB attn (A op)
  __shared__ __align__(16) _Float16 sW2T[128 * 128];      // 32 KB W2^T (B op)
  __shared__ float sSi[64], sSj[64];
  __shared__ float sVec[128];
  __shared__ float sR1[32];
  __shared__ float sRawD[64], sRawP[64];
  __shared__ float sSum[2];

  const int b    = blockIdx.x;
  const int tid  = threadIdx.x;
  const int lane = tid & 31;
  const int wave = tid >> 5;

  // Prefetch the late-phase readout weights toward the caches.
  __builtin_prefetch(mw1, 0, 2);
  __builtin_prefetch(mw2, 0, 2);

#if GAT_HAVE_TDM
  // Tensor Data Mover: DMA W2 (64 KB) and this block's users tile into LDS.
  // Wave 0 issues both (TDM ignores EXEC; TENSORcnt is per-wave), waits,
  // then the workgroup barrier publishes the LDS to all waves.
  if (wave == 0) {
    tdm_load_1d_f32(W2, sW2stage, 128 * 128);
    tdm_load_1d_f32(users + (size_t)b * 64 * 3, sUsers, 64 * 3);
    __builtin_amdgcn_s_wait_tensorcnt(0);
  }
#else
  for (int i = tid; i < 128 * 128; i += 256) sW2stage[i] = W2[i];
  for (int i = tid; i < 64 * 3; i += 256)
    sUsers[i] = users[(size_t)b * 64 * 3 + i];
#endif
  __syncthreads();

  // Convert staged W2 to f16 TRANSPOSED (B operand of layer-2 GEMM).
  for (int i = tid; i < 128 * 128; i += 256) {
    const int k = i >> 7, n = i & 127;
    sW2T[n * 128 + k] = (_Float16)sW2stage[i];
  }

  // Wh1 = users(64x3) @ W1(3x128), stored transposed as f16 (B operand).
  for (int i = tid; i < 64 * 128; i += 256) {
    const int m = i >> 7, d = i & 127;
    const float* u = sUsers + m * 3;
    const float v =
        fmaf(u[0], W1[d], fmaf(u[1], W1[128 + d], u[2] * W1[256 + d]));
    sWhT[d * 64 + m] = (_Float16)v;
  }
  __syncthreads();

  // ---- GAT layer 1: h1 = elu(attn1 @ Wh1) -> sHo (row-major f16) ----
  gat_attention(a1, sWhT, sHo, sAt, sSi, sSj, tid, lane, wave);

  // ---- Layer 2 projection: Wh2 = h1(64x128) @ W2(128x128) ----
  // A = sHo (row-major), B = sW2T; result stored TRANSPOSED into sWhT
  // (one ds_store_b128 per tile-lane) since attn-2 consumes it as B.
  {
    const int n0 = wave * 16;
    v16h bf[4];
#pragma unroll
    for (int kt = 0; kt < 4; ++kt)
      bf[kt] = frag_row_f16(sW2T, 128, n0, kt * 32, lane);
#pragma unroll
    for (int mt = 0; mt < 4; ++mt) {
      v8f c = {};
#pragma unroll
      for (int kt = 0; kt < 4; ++kt) {
        v16h a = frag_row_f16(sHo, 128, mt * 16, kt * 32, lane);
        c = __builtin_amdgcn_wmma_f32_16x16x32_f16(false, a, false, bf[kt],
                                                   (short)0, c, false, false);
      }
      store_c_tr_f16(sWhT, 64, mt * 16, n0, lane, c);
    }
  }
  __syncthreads();

  // ---- GAT layer 2: h2 = elu(attn2 @ Wh2) -> sHo ----
  gat_attention(a2, sWhT, sHo, sAt, sSi, sSj, tid, lane, wave);

  // ---- pooled = max over nodes ----
  if (tid < 128) {
    float mx = (float)sHo[tid];
    for (int m = 1; m < 64; ++m) mx = fmaxf(mx, (float)sHo[m * 128 + tid]);
    sVec[tid] = mx;
  }
  __syncthreads();

  // ---- readout MLP: relu(pooled @ mw1 + mb1) @ mw2 + mb2 ----
  if (tid < 32) {
    float s = mb1[tid];
    for (int d = 0; d < 128; ++d) s = fmaf(sVec[d], mw1[d * 32 + tid], s);
    sR1[tid] = fmaxf(s, 0.f);
  }
  __syncthreads();
  if (tid < 128) {
    float s = mb2[tid];
    for (int k = 0; k < 32; ++k) s = fmaf(sR1[k], mw2[k * 128 + tid], s);
    sVec[tid] = s;
  }
  __syncthreads();

  // ---- split, relu+eps, normalize ----
  if (tid < 64) {
    sRawD[tid] = fmaxf(sVec[tid], 0.f) + 1e-6f;
    sRawP[tid] = fmaxf(sVec[64 + tid], 0.f) + 1e-6f;
  }
  __syncthreads();
  if (tid == 0) {
    float sd = 0.f, sp = 0.f;
    for (int j = 0; j < 64; ++j) { sd += sRawD[j]; sp += sRawP[j]; }
    sSum[0] = sd;
    sSum[1] = sp;
  }
  __syncthreads();
  if (tid < 64) {
    // DELTA_MIN = 3e8/6e9/2 = 0.025 ; Bmax = 100 - 63*0.025 ; PMAX = 1.0
    const float Bmax = 2.0f * 50.0f - 63.0f * 0.025f;
    out_power[(size_t)b * 64 + tid] = 1.0f * sRawP[tid] / (sSum[1] + 1e-6f);
    out_delta[(size_t)b * 64 + tid] = Bmax * sRawD[tid] / (sSum[0] + 1e-6f);
  }
}

extern "C" void kernel_launch(void* const* d_in, const int* in_sizes, int n_in,
                              void* d_out, int out_size, void* d_ws,
                              size_t ws_size, hipStream_t stream) {
  const float* users = (const float*)d_in[0];
  const float* W1    = (const float*)d_in[1];
  const float* a1    = (const float*)d_in[2];
  const float* W2    = (const float*)d_in[3];
  const float* a2    = (const float*)d_in[4];
  const float* mw1   = (const float*)d_in[5];
  const float* mb1   = (const float*)d_in[6];
  const float* mw2   = (const float*)d_in[7];
  const float* mb2   = (const float*)d_in[8];

  const int B = in_sizes[0] / (64 * 3);   // 4096
  float* outp      = (float*)d_out;
  float* out_power = outp;                     // first tuple element (power)
  float* out_delta = outp + (size_t)B * 64;    // second tuple element (delta)

  gat_fused_kernel<<<B, 256, 0, stream>>>(users, W1, a1, W2, a2, mw1, mb1,
                                          mw2, mb2, out_power, out_delta);
}
